// Head_6493990551853
// MI455X (gfx1250) — matrix operand order repaired and checked by reading, not compiled, not run on hardware
//
#include <hip/hip_runtime.h>
#include <stdint.h>

#define Bsz  8
#define Tseq 4096
#define Cdim 1024
#define Hdim 128

typedef __attribute__((ext_vector_type(16))) __bf16   v16bf;
typedef __attribute__((ext_vector_type(8)))  float    v8f;
typedef __attribute__((ext_vector_type(4)))  uint32_t u32x4;
typedef __attribute__((ext_vector_type(4)))  float    f32x4;

union BF16Frag {
    v16bf          v;
    __bf16         h[16];
    unsigned short u[16];
    u32x4          q[2];
};

union BF16Pack8 {                 // 8 bf16 = one 16B store
    u32x4  q;
    __bf16 h[8];
};

// ---------------------------------------------------------------------------
// Kernel 0: Wq/Wk/Wv f32 [C][H]  ->  bf16 transposed [3][H][C]
// ---------------------------------------------------------------------------
__global__ void wtrans(const float* __restrict__ wq, const float* __restrict__ wk,
                       const float* __restrict__ wv, __bf16* __restrict__ wt) {
    int idx = blockIdx.x * 256 + threadIdx.x;
    if (idx >= 3 * Hdim * Cdim) return;
    int which = idx / (Hdim * Cdim);
    int rem   = idx % (Hdim * Cdim);
    int h = rem / Cdim, c = rem % Cdim;
    const float* w = (which == 0) ? wq : ((which == 1) ? wk : wv);
    wt[idx] = (__bf16)w[c * Hdim + h];
}

// ---------------------------------------------------------------------------
// Kernel 1: QKV projection.  grid = (T/128, B, 3), block = 128 (4 waves).
// Each wave: 32x128 output (two 16-row tiles sharing the B fragments).
// Q is pre-scaled by log2(e)/sqrt(H): the attention softmax then runs entirely
// in the exp2 domain (one v_exp_f32 per exponential, no log2e multiplies).
// Q,K stored [B][T][H]; V stored transposed [B][H][T].
// ---------------------------------------------------------------------------
__global__ __launch_bounds__(128) void qkv_proj(
    const float* __restrict__ x, const __bf16* __restrict__ wt,
    __bf16* __restrict__ qout, __bf16* __restrict__ kout,
    __bf16* __restrict__ vtout) {

    const int which = blockIdx.z;
    const int b     = blockIdx.y;
    const int wave  = threadIdx.x >> 5;
    const int lane  = threadIdx.x & 31;
    const int lrow  = lane & 15;
    const bool hi   = lane >= 16;
    const int m0    = blockIdx.x * 128 + wave * 32;     // 32 rows per wave
    const int kbA   = hi ? 8 : 0;    // A-frag K base (16-bit A layout)
    const int kbB   = hi ? 8 : 0;    // B-frag K base (16-bit B layout)

    const float*  xrow0 = x + ((size_t)b * Tseq + (m0 + lrow)) * Cdim;
    const float*  xrow1 = xrow0 + (size_t)16 * Cdim;
    const __bf16* wbase = wt + (size_t)which * Hdim * Cdim;

    v8f acc0[8] = {};
    v8f acc1[8] = {};

    for (int kk = 0; kk < Cdim; kk += 32) {
        // ---- preload all 8 B fragments for this k-step
        BF16Frag bw[8];
#pragma unroll
        for (int n = 0; n < 8; n++) {
            const __bf16* wp = wbase + (size_t)(n * 16 + lrow) * Cdim + kk + kbB;
            bw[n].q[0] = *(const u32x4*)(wp);
            bw[n].q[1] = *(const u32x4*)(wp + 16);
        }
        // ---- A fragments for both row tiles (f32 -> bf16, native cvt)
        BF16Frag a0, a1;
        {
            f32x4 f0 = *(const f32x4*)(xrow0 + kk + kbA);
            f32x4 f1 = *(const f32x4*)(xrow0 + kk + kbA + 4);
            f32x4 f2 = *(const f32x4*)(xrow0 + kk + kbA + 16);
            f32x4 f3 = *(const f32x4*)(xrow0 + kk + kbA + 20);
            f32x4 g0 = *(const f32x4*)(xrow1 + kk + kbA);
            f32x4 g1 = *(const f32x4*)(xrow1 + kk + kbA + 4);
            f32x4 g2 = *(const f32x4*)(xrow1 + kk + kbA + 16);
            f32x4 g3 = *(const f32x4*)(xrow1 + kk + kbA + 20);
#pragma unroll
            for (int i = 0; i < 4; i++) {
                a0.h[i]      = (__bf16)f0[i];
                a0.h[4 + i]  = (__bf16)f1[i];
                a0.h[8 + i]  = (__bf16)f2[i];
                a0.h[12 + i] = (__bf16)f3[i];
                a1.h[i]      = (__bf16)g0[i];
                a1.h[4 + i]  = (__bf16)g1[i];
                a1.h[8 + i]  = (__bf16)g2[i];
                a1.h[12 + i] = (__bf16)g3[i];
            }
        }
        // ---- 16 WMMAs reusing the preloaded B fragments
#pragma unroll
        for (int n = 0; n < 8; n++) {
            acc0[n] = __builtin_amdgcn_wmma_f32_16x16x32_bf16(
                false, a0.v, false, bw[n].v, (short)0, acc0[n], false, false);
            acc1[n] = __builtin_amdgcn_wmma_f32_16x16x32_bf16(
                false, a1.v, false, bw[n].v, (short)0, acc1[n], false, false);
        }
    }

    // fold softmax scale AND log2(e) into Q:  log2(e)/sqrt(128)
    const float sc = (which == 0) ? 0.1275187929081774f : 1.0f;

    if (which < 2) {
        __bf16* dst = ((which == 0) ? qout : kout) + (size_t)b * Tseq * Hdim;
#pragma unroll
        for (int n = 0; n < 8; n++)
#pragma unroll
            for (int r = 0; r < 8; r++) {
                int row0 = m0 + r + (hi ? 8 : 0);
                dst[(size_t)row0 * Hdim + n * 16 + lrow]        = (__bf16)(acc0[n][r] * sc);
                dst[(size_t)(row0 + 16) * Hdim + n * 16 + lrow] = (__bf16)(acc1[n][r] * sc);
            }
    } else {
        // transposed V: rows r are contiguous along T -> packed 16B stores
        __bf16* dst = vtout + (size_t)b * Hdim * Tseq;   // [H][T]
#pragma unroll
        for (int n = 0; n < 8; n++) {
            BF16Pack8 p0, p1;
#pragma unroll
            for (int r = 0; r < 8; r++) {
                p0.h[r] = (__bf16)acc0[n][r];
                p1.h[r] = (__bf16)acc1[n][r];
            }
            __bf16* base = dst + (size_t)(n * 16 + lrow) * Tseq + m0 + (hi ? 8 : 0);
            *(u32x4*)(base)      = p0.q;
            *(u32x4*)(base + 16) = p1.q;
        }
    }
}

// ---------------------------------------------------------------------------
// online-softmax update (log2 domain) for one 16-row query tile over a 32-key
// block.  Scores arrive pre-multiplied by log2(e)/sqrt(H); all exponentials
// are exp2f -> single v_exp_f32 each.  Writes exp2(P) bf16 into LDS staging.
// ---------------------------------------------------------------------------
__device__ __forceinline__ void tile_update(
    v8f sA, v8f sB, bool diagB,
    int qbase, int kA0, int lrow, bool hi,
    float* mrun, float* lrun, v8f* o, __bf16* Pb)
{
    const float NEG_INF = -__builtin_inff();
#pragma unroll
    for (int r = 0; r < 8; r++) {
        int row = qbase + r + (hi ? 8 : 0);
        float a  = sA[r];
        float bb = sB[r];
        if (diagB && (kA0 + 16 + lrow > row)) bb = NEG_INF;
        float lm = fmaxf(a, bb);
#pragma unroll
        for (int off = 1; off < 16; off <<= 1)
            lm = fmaxf(lm, __shfl_xor(lm, off, 16));
        float mn = fmaxf(mrun[r], lm);
        float al = exp2f(mrun[r] - mn);             // 0 when mrun == -inf
        float pa = exp2f(a - mn);
        float pb = exp2f(bb - mn);
        float rs = pa + pb;
#pragma unroll
        for (int off = 1; off < 16; off <<= 1)
            rs += __shfl_xor(rs, off, 16);
        lrun[r] = lrun[r] * al + rs;
        mrun[r] = mn;
        int prow = r + (hi ? 8 : 0);
        Pb[prow * 32 + lrow]      = (__bf16)pa;
        Pb[prow * 32 + lrow + 16] = (__bf16)pb;
#pragma unroll
        for (int n = 0; n < 8; n++) o[n][r] *= al;
    }
}

// ---------------------------------------------------------------------------
// Kernel 2: causal flash attention. grid = (T/32, B), block = 32 (one wave).
// 32 query rows per wave (two 16-row tiles) -> every K/V fragment load feeds
// two WMMAs.  Uniform loop body: the second score half is always computed
// (rows stay in-bounds) and killed with -inf when past the causal frontier.
// ---------------------------------------------------------------------------
__global__ __launch_bounds__(32) void attn(
    const __bf16* __restrict__ q, const __bf16* __restrict__ k,
    const __bf16* __restrict__ vt, float* __restrict__ out) {

    __shared__ __bf16 Pbuf0[16 * 32];
    __shared__ __bf16 Pbuf1[16 * 32];

    const int  b    = blockIdx.y;
    const int  q0   = blockIdx.x * 32;      // rows q0 .. q0+31
    const int  lane = threadIdx.x & 31;
    const int  lrow = lane & 15;
    const bool hi   = lane >= 16;
    const int  kbA  = hi ? 8 : 0;
    const int  kbB  = hi ? 8 : 0;
    const float NEG_INF = -__builtin_inff();

    const __bf16* qb = q  + (size_t)b * Tseq * Hdim;
    const __bf16* kb = k  + (size_t)b * Tseq * Hdim;
    const __bf16* vb = vt + (size_t)b * Hdim * Tseq;

    // Q A-fragments for both query tiles (Q pre-scaled by log2(e)/sqrt(H))
    BF16Frag qa0[4], qa1[4];
    {
        const __bf16* qr0 = qb + (size_t)(q0 + lrow) * Hdim;
        const __bf16* qr1 = qr0 + (size_t)16 * Hdim;
#pragma unroll
        for (int j = 0; j < 4; j++) {
            qa0[j].q[0] = *(const u32x4*)(qr0 + j * 32 + kbA);
            qa0[j].q[1] = *(const u32x4*)(qr0 + j * 32 + kbA + 16);
            qa1[j].q[0] = *(const u32x4*)(qr1 + j * 32 + kbA);
            qa1[j].q[1] = *(const u32x4*)(qr1 + j * 32 + kbA + 16);
        }
    }

    v8f o0[8] = {}, o1[8] = {};
    float m0r[8], l0r[8], m1r[8], l1r[8];
#pragma unroll
    for (int r = 0; r < 8; r++) {
        m0r[r] = NEG_INF; l0r[r] = 0.f;
        m1r[r] = NEG_INF; l1r[r] = 0.f;
    }

    for (int key0 = 0; key0 <= q0; key0 += 32) {
        // speculative prefetch of the next key block (K rows + transposed V)
        if (key0 + 32 <= q0) {
            __builtin_prefetch(kb + (size_t)(key0 + 32 + lrow) * Hdim, 0, 1);
            __builtin_prefetch(vb + (size_t)(lane * 4) * Tseq + key0 + 32, 0, 1);
        }

        // ---- K fragments for keys [key0, key0+16); score tiles for both q-tiles
        v8f sA0 = {}, sA1 = {};
        {
            BF16Frag kf[4];
#pragma unroll
            for (int j = 0; j < 4; j++) {
                const __bf16* kp = kb + (size_t)(key0 + lrow) * Hdim + j * 32 + kbB;
                kf[j].q[0] = *(const u32x4*)kp;
                kf[j].q[1] = *(const u32x4*)(kp + 16);
            }
#pragma unroll
            for (int j = 0; j < 4; j++) {
                sA0 = __builtin_amdgcn_wmma_f32_16x16x32_bf16(
                    false, qa0[j].v, false, kf[j].v, (short)0, sA0, false, false);
                sA1 = __builtin_amdgcn_wmma_f32_16x16x32_bf16(
                    false, qa1[j].v, false, kf[j].v, (short)0, sA1, false, false);
            }
        }

        // ---- K fragments for keys [key0+16, key0+32); always in-bounds
        const int k0b = key0 + 16;
        v8f sB0 = {}, sB1 = {};
        {
            BF16Frag kf[4];
#pragma unroll
            for (int j = 0; j < 4; j++) {
                const __bf16* kp = kb + (size_t)(k0b + lrow) * Hdim + j * 32 + kbB;
                kf[j].q[0] = *(const u32x4*)kp;
                kf[j].q[1] = *(const u32x4*)(kp + 16);
            }
#pragma unroll
            for (int j = 0; j < 4; j++) {
                sB0 = __builtin_amdgcn_wmma_f32_16x16x32_bf16(
                    false, qa0[j].v, false, kf[j].v, (short)0, sB0, false, false);
                sB1 = __builtin_amdgcn_wmma_f32_16x16x32_bf16(
                    false, qa1[j].v, false, kf[j].v, (short)0, sB1, false, false);
            }
        }
        // tile0: kill the B half entirely when it lies past the causal frontier
        if (k0b > q0) {
#pragma unroll
            for (int r = 0; r < 8; r++) sB0[r] = NEG_INF;
        }

        // Note on diagonals: key0 <= q0, both multiples of 32/16 =>
        //   tile0 A-half never masked except via the kill above (key0 == q0
        //   only at the final block, where A covers cols <= row);  actually
        //   key0 == q0 makes A the diagonal of tile0 -> handled as diagB of
        //   the *previous* 16 offset; here A's diagonal test:
        //   cols key0+lrow vs rows q0+...  key0 == q0 needs masking on A.
        if (key0 == q0) {
#pragma unroll
            for (int r = 0; r < 8; r++) {
                int row = q0 + r + (hi ? 8 : 0);
                if (key0 + lrow > row) sA0[r] = NEG_INF;
            }
        }

        tile_update(sA0, sB0, /*diagB=*/(k0b == q0),
                    q0, key0, lrow, hi, m0r, l0r, o0, Pbuf0);
        tile_update(sA1, sB1, /*diagB=*/(key0 == q0),   // k0b == q0+16
                    q0 + 16, key0, lrow, hi, m1r, l1r, o1, Pbuf1);
        __syncthreads();

        // ---- P (C-layout) -> A-fragments via LDS
        BF16Frag pf0, pf1;
        pf0.q[0] = *(const u32x4*)(&Pbuf0[lrow * 32 + kbA]);
        pf0.q[1] = *(const u32x4*)(&Pbuf0[lrow * 32 + kbA + 16]);
        pf1.q[0] = *(const u32x4*)(&Pbuf1[lrow * 32 + kbA]);
        pf1.q[1] = *(const u32x4*)(&Pbuf1[lrow * 32 + kbA + 16]);

        // ---- O += P x V; each V fragment feeds both query tiles
#pragma unroll
        for (int g = 0; g < 4; g++) {
            BF16Frag vf[2];
#pragma unroll
            for (int t = 0; t < 2; t++) {
                int n = g * 2 + t;
                const __bf16* vp = vb + (size_t)(n * 16 + lrow) * Tseq + key0 + kbB;
                vf[t].q[0] = *(const u32x4*)vp;
                vf[t].q[1] = *(const u32x4*)(vp + 16);
            }
#pragma unroll
            for (int t = 0; t < 2; t++) {
                int n = g * 2 + t;
                o0[n] = __builtin_amdgcn_wmma_f32_16x16x32_bf16(
                    false, pf0.v, false, vf[t].v, (short)0, o0[n], false, false);
                o1[n] = __builtin_amdgcn_wmma_f32_16x16x32_bf16(
                    false, pf1.v, false, vf[t].v, (short)0, o1[n], false, false);
            }
        }
        __syncthreads();
    }

    // ---- epilogue: normalize and store f32 for both tiles
    float* ob = out + (size_t)b * Tseq * Hdim;
#pragma unroll
    for (int r = 0; r < 8; r++) {
        int   row0 = q0 + r + (hi ? 8 : 0);
        float inv0 = 1.0f / l0r[r];
        float inv1 = 1.0f / l1r[r];
#pragma unroll
        for (int n = 0; n < 8; n++) {
            ob[(size_t)row0 * Hdim + n * 16 + lrow]        = o0[n][r] * inv0;
            ob[(size_t)(row0 + 16) * Hdim + n * 16 + lrow] = o1[n][r] * inv1;
        }
    }
}

// ---------------------------------------------------------------------------
extern "C" void kernel_launch(void* const* d_in, const int* in_sizes, int n_in,
                              void* d_out, int out_size, void* d_ws, size_t ws_size,
                              hipStream_t stream) {
    const float* x  = (const float*)d_in[0];
    const float* Wq = (const float*)d_in[1];
    const float* Wk = (const float*)d_in[2];
    const float* Wv = (const float*)d_in[3];
    float* out = (float*)d_out;

    char* ws = (char*)d_ws;
    __bf16* wt = (__bf16*)ws;                        // [3][H][C] bf16 (~768KB)
    __bf16* Q  = (__bf16*)(ws + (1u << 20));         // [B][T][H] bf16 (8MB)
    __bf16* K  = Q + (size_t)Bsz * Tseq * Hdim;      // [B][T][H] bf16 (8MB)
    __bf16* Vt = K + (size_t)Bsz * Tseq * Hdim;      // [B][H][T] bf16 (8MB)

    {   // weights -> bf16 transposed
        int total = 3 * Hdim * Cdim;
        wtrans<<<(total + 255) / 256, 256, 0, stream>>>(Wq, Wk, Wv, wt);
    }
    {   // QKV projection (Q pre-scaled by log2(e)/sqrt(H))
        dim3 g(Tseq / 128, Bsz, 3);
        qkv_proj<<<g, 128, 0, stream>>>(x, wt, Q, K, Vt);
    }
    {   // causal flash attention, 32 query rows per wave
        dim3 g(Tseq / 32, Bsz);
        attn<<<g, 32, 0, stream>>>(Q, K, Vt, out);
    }
}
